// STRPN_81217831567849
// MI455X (gfx1250) — matrix operand order repaired and checked by reading, not compiled, not run on hardware
//
#include <hip/hip_runtime.h>
#include <hip/hip_fp16.h>

typedef __attribute__((ext_vector_type(16))) _Float16 v16h;
typedef __attribute__((ext_vector_type(8)))  _Float16 v8h;
typedef __attribute__((ext_vector_type(8)))  float    v8f;

#define FH 96
#define FW 96
#define NPIX (FH*FW)            // 9216
#define CIN 1024
#define CMID 512
#define NA 9
#define NANCH (NPIX*NA)         // 82944
#define PRE_N 12000
#define POST_N 300
#define SORT_N 131072
#define POOL 7

// ---------------------------------------------------------------------------
// Weight packing: OIHW f32 -> WMMA B-fragment order in f16, laid out as
// (tap, kchunk32, ntile16) so the 4 fragments a wave needs per (tap,kchunk)
// are 4 adjacent 1KB blocks.  Within a 512-half block: lane L<16 holds N=L,
// K=j (j=0..15); lane L>=16 holds N=L-16, K=16+j.
// ---------------------------------------------------------------------------
__global__ void k_pack_w(const float* __restrict__ w, _Float16* __restrict__ wpk) {
    int i = blockIdx.x * 256 + threadIdx.x;
    if (i >= 9 * CMID * CIN) return;
    int tap = i / (CMID * CIN);
    int rem = i - tap * (CMID * CIN);
    int n = rem / CIN;
    int c = rem - n * CIN;
    float v = w[(size_t)(n * CIN + c) * 9 + tap];
    int kl = c & 31;
    int L  = (kl < 16) ? (n & 15) : (16 + (n & 15));
    int j  = kl & 15;
    int blk = (tap * (CIN / 32) + (c >> 5)) * (CMID / 16) + (n >> 4);
    wpk[(size_t)blk * 512 + L * 16 + j] = (_Float16)v;
}

// Load one tap's fragments: A from LDS (2 x b128), B0..B3 from packed global.
__device__ __forceinline__ void loadfr(const _Float16* __restrict__ buf, unsigned abase,
                                       const _Float16* __restrict__ wpk, unsigned bb, int tap,
                                       v16h& A, v16h& B0, v16h& B1, v16h& B2, v16h& B3) {
    int dy = tap / 3, dx = tap - dy * 3;
    const _Float16* pa = buf + (unsigned)((dy * 34 + dx) * 32) + abase;
    v8h alo = *(const v8h*)pa;          // K base..base+7
    v8h ahi = *(const v8h*)(pa + 8);    // K base+8..base+15
    #pragma unroll
    for (int j = 0; j < 8; ++j) { A[j] = alo[j]; A[8 + j] = ahi[j]; }
    const _Float16* pb = wpk + bb + (unsigned)tap * 524288u;   // tap stride = 32*32 blocks
    B0 = *(const v16h*)(pb);
    B1 = *(const v16h*)(pb + 512);
    B2 = *(const v16h*)(pb + 1024);
    B3 = *(const v16h*)(pb + 1536);
}

// ---------------------------------------------------------------------------
// 3x3 SAME conv + bias + ReLU via implicit GEMM over filter taps.
// 8 waves; each wave computes FOUR 16x16 output tiles (4 independent WMMA
// accumulator chains, A reused 4x).  Double-buffered LDS staging (1 barrier
// per k-chunk) + one-tap software pipeline on the A/B fragments.
// Workgroup tile: 32 pixels x 256 channels.  Grid: (96, 3, 2).
// ---------------------------------------------------------------------------
__global__ __launch_bounds__(256) void k_conv3x3(const float* __restrict__ x,
                                                 const _Float16* __restrict__ wpk,
                                                 const float* __restrict__ bias,
                                                 _Float16* __restrict__ out) {
    const int y    = blockIdx.x;
    const int x0   = blockIdx.y * 32;
    const int tid  = threadIdx.x;
    const int lane = tid & 31;
    const int wave = tid >> 5;
    const int msub = wave & 1;
    const int nsub = wave >> 1;                    // 0..3
    const int m    = lane & 15;
    const int hi   = lane >> 4;
    const int ntbase = blockIdx.z * 16 + nsub * 4; // global 16-wide N-tile id

    __shared__ __align__(16) _Float16 sA[2][3 * 34 * 32];   // double buffer

    v8f acc0 = {0.f,0.f,0.f,0.f,0.f,0.f,0.f,0.f};
    v8f acc1 = acc0, acc2 = acc0, acc3 = acc0;

    const unsigned abase  = (unsigned)((msub * 16 + m) * 32 + hi * 16);
    const unsigned bbase0 = (unsigned)ntbase * 512u + (unsigned)lane * 16u;

    // Stage 3 rows x 34 cols x 32 channels (halo, zero pad) as f16,
    // prefetching the following chunk's lines into cache.
    auto stage = [&](int c0, _Float16* buf) {
        for (int i = tid; i < 3 * 34 * 32; i += 256) {
            int ch  = i / 102;
            int rem = i - ch * 102;
            int r   = rem / 34;
            int col = rem - r * 34;
            int yy = y + r - 1;
            int xx = x0 + col - 1;
            float v = 0.f;
            if ((unsigned)yy < FH && (unsigned)xx < FW) {
                const float* gp = &x[(c0 + ch) * NPIX + yy * FW + xx];
                v = *gp;
                if (c0 + 32 < CIN) __builtin_prefetch(gp + 32 * NPIX, 0, 0);
            }
            buf[(r * 34 + col) * 32 + ch] = (_Float16)v;
        }
    };

    stage(0, sA[0]);
    __syncthreads();

    for (int kc = 0; kc < CIN / 32; ++kc) {
        const _Float16* buf = sA[kc & 1];
        if (kc + 1 < CIN / 32)
            stage((kc + 1) * 32, sA[(kc & 1) ^ 1]);   // overlaps with compute below

        const unsigned bb = bbase0 + (unsigned)kc * 16384u;   // kc stride = 32 blocks

        v16h A, B0, B1, B2, B3, An, B0n, B1n, B2n, B3n;
        loadfr(buf, abase, wpk, bb, 0, A, B0, B1, B2, B3);
        #pragma unroll
        for (int tap = 0; tap < 9; ++tap) {
            if (tap < 8)
                loadfr(buf, abase, wpk, bb, tap + 1, An, B0n, B1n, B2n, B3n);
            acc0 = __builtin_amdgcn_wmma_f32_16x16x32_f16(false, A, false, B0, (short)0, acc0, false, false);
            acc1 = __builtin_amdgcn_wmma_f32_16x16x32_f16(false, A, false, B1, (short)0, acc1, false, false);
            acc2 = __builtin_amdgcn_wmma_f32_16x16x32_f16(false, A, false, B2, (short)0, acc2, false, false);
            acc3 = __builtin_amdgcn_wmma_f32_16x16x32_f16(false, A, false, B3, (short)0, acc3, false, false);
            if (tap < 8) { A = An; B0 = B0n; B1 = B1n; B2 = B2n; B3 = B3n; }
        }
        __syncthreads();
    }

    // Epilogue: C layout lane<16 -> N=lane, M=v ; lane>=16 -> M=8+v.
    const int xbase = x0 + msub * 16 + (hi ? 8 : 0);
    v8f accs[4] = {acc0, acc1, acc2, acc3};
    #pragma unroll
    for (int q = 0; q < 4; ++q) {
        int n = (ntbase + q) * 16 + m;
        float b = bias[n];
        #pragma unroll
        for (int v = 0; v < 8; ++v) {
            float val = accs[q][v] + b;
            val = val > 0.f ? val : 0.f;
            out[n * NPIX + y * FW + (xbase + v)] = (_Float16)val;
        }
    }
}

// ---------------------------------------------------------------------------
// 1x1 conv over f16 activations (0.6% of conv3x3 FLOPs -> VALU is fine).
// ---------------------------------------------------------------------------
__global__ void k_conv1x1(const _Float16* __restrict__ in, const float* __restrict__ w,
                          const float* __restrict__ b, float* __restrict__ out) {
    int p  = blockIdx.x * 256 + threadIdx.x;
    int nc = blockIdx.y;
    float acc = b[nc];
    const float* wr = w + nc * CMID;
    for (int c = 0; c < CMID; ++c)
        acc += (float)in[c * NPIX + p] * wr[c];
    out[nc * NPIX + p] = acc;
}

// fg = softmax over the 2-class axis, flattened in (h,w,a) order.
__global__ void k_scores(const float* __restrict__ cls, float* __restrict__ scores) {
    int i = blockIdx.x * 256 + threadIdx.x;
    int p = i / NA, a = i - p * NA;
    float l0 = cls[a * NPIX + p];
    float l1 = cls[(NA + a) * NPIX + p];
    float mx = fmaxf(l0, l1);
    float e0 = expf(l0 - mx), e1 = expf(l1 - mx);
    scores[i] = e1 / (e0 + e1);
}

// Anchors (rintf == np.round half-even), decode deltas, clip.
__global__ void k_anch_decode(const float* __restrict__ bbox, const float* __restrict__ iminfo,
                              float* __restrict__ anch, float* __restrict__ prop) {
    int i = blockIdx.x * 256 + threadIdx.x;
    int p = i / NA, a = i - p * NA;
    int h = p / FW, w = p - h * FW;
    int ri = a / 3, si = a - ri * 3;
    float ratio = (ri == 0) ? 0.5f : ((ri == 1) ? 1.f : 2.f);
    float scale = (si == 0) ? 8.f : ((si == 1) ? 16.f : 32.f);
    float wsr = rintf(sqrtf(256.f / ratio));
    float hsr = rintf(wsr * ratio);
    float ws = wsr * scale, hs = hsr * scale;
    float sx = w * 16.f, sy = h * 16.f;
    float ax1 = sx + 7.5f - 0.5f * (ws - 1.f);
    float ay1 = sy + 7.5f - 0.5f * (hs - 1.f);
    float ax2 = sx + 7.5f + 0.5f * (ws - 1.f);
    float ay2 = sy + 7.5f + 0.5f * (hs - 1.f);
    anch[i * 4 + 0] = ax1; anch[i * 4 + 1] = ay1;
    anch[i * 4 + 2] = ax2; anch[i * 4 + 3] = ay2;
    float aw = ax2 - ax1 + 1.f, ah = ay2 - ay1 + 1.f;
    float acx = ax1 + 0.5f * aw, acy = ay1 + 0.5f * ah;
    float d0 = bbox[(a * 4 + 0) * NPIX + p], d1 = bbox[(a * 4 + 1) * NPIX + p];
    float d2 = bbox[(a * 4 + 2) * NPIX + p], d3 = bbox[(a * 4 + 3) * NPIX + p];
    float pcx = d0 * aw + acx, pcy = d1 * ah + acy;
    float pw = expf(d2) * aw, ph = expf(d3) * ah;
    float imh = iminfo[0], imw = iminfo[1];
    prop[i * 4 + 0] = fminf(fmaxf(pcx - 0.5f * pw, 0.f), imw - 1.f);
    prop[i * 4 + 1] = fminf(fmaxf(pcy - 0.5f * ph, 0.f), imh - 1.f);
    prop[i * 4 + 2] = fminf(fmaxf(pcx + 0.5f * pw, 0.f), imw - 1.f);
    prop[i * 4 + 3] = fminf(fmaxf(pcy + 0.5f * ph, 0.f), imh - 1.f);
}

// ---------------------------------------------------------------------------
// top_k via bitonic sort of (~orderedScore, idx) keys (stable descending).
// ---------------------------------------------------------------------------
__device__ __forceinline__ unsigned orderf(float f) {
    unsigned u = __float_as_uint(f);
    return (u & 0x80000000u) ? ~u : (u | 0x80000000u);
}
__global__ void k_init_keys(const float* __restrict__ scores,
                            unsigned long long* __restrict__ keys) {
    int i = blockIdx.x * 256 + threadIdx.x;
    if (i >= SORT_N) return;
    keys[i] = (i < NANCH)
            ? (((unsigned long long)(~orderf(scores[i])) << 32) | (unsigned)i)
            : ~0ULL;
}
__global__ void k_bitonic(unsigned long long* keys, int k, int j) {
    int i = blockIdx.x * 256 + threadIdx.x;
    int ixj = i ^ j;
    if (ixj <= i) return;
    unsigned long long a = keys[i], b = keys[ixj];
    bool asc = ((i & k) == 0);
    if (asc ? (a > b) : (a < b)) { keys[i] = b; keys[ixj] = a; }
}
__global__ void k_gather_top(const unsigned long long* __restrict__ keys,
                             const float* __restrict__ prop, float* __restrict__ props) {
    int i = blockIdx.x * 256 + threadIdx.x;
    if (i >= PRE_N) return;
    int idx = (int)(keys[i] & 0xFFFFFFFFu);
    props[i * 4 + 0] = prop[idx * 4 + 0];
    props[i * 4 + 1] = prop[idx * 4 + 1];
    props[i * 4 + 2] = prop[idx * 4 + 2];
    props[i * 4 + 3] = prop[idx * 4 + 3];
}

// ---------------------------------------------------------------------------
// Greedy NMS, single workgroup, bitmask in LDS.  Thread t owns word t ->
// no atomics, deterministic.  375 words * 32 = 12000 exactly.
// ---------------------------------------------------------------------------
__global__ __launch_bounds__(512) void k_nms(const float* __restrict__ props,
                                             int* __restrict__ keep) {
    __shared__ unsigned valid[375];
    __shared__ int red[512];
    __shared__ float box[4];
    int t = threadIdx.x;
    if (t < 375) valid[t] = 0xFFFFFFFFu;
    __syncthreads();
    for (int it = 0; it < POST_N; ++it) {
        int f = 0x7FFFFFFF;
        if (t < 375 && valid[t]) f = t * 32 + __builtin_ctz(valid[t]);
        red[t] = f;
        __syncthreads();
        for (int s = 256; s > 0; s >>= 1) {
            if (t < s) red[t] = min(red[t], red[t + s]);
            __syncthreads();
        }
        int idx = red[0];
        bool has = idx < PRE_N;
        if (t == 0) {
            keep[it] = has ? idx : 0;
            if (has) {
                box[0] = props[idx * 4 + 0]; box[1] = props[idx * 4 + 1];
                box[2] = props[idx * 4 + 2]; box[3] = props[idx * 4 + 3];
            }
        }
        __syncthreads();
        if (has && t < 375) {
            unsigned v = valid[t];
            if (v) {
                float bx1 = box[0], by1 = box[1], bx2 = box[2], by2 = box[3];
                float barea = (bx2 - bx1 + 1.f) * (by2 - by1 + 1.f);
                unsigned nv = v;
                for (int b = 0; b < 32; ++b) {
                    if (!((v >> b) & 1u)) continue;
                    int q = t * 32 + b;
                    float x1 = props[q * 4 + 0], y1 = props[q * 4 + 1];
                    float x2 = props[q * 4 + 2], y2 = props[q * 4 + 3];
                    float xx1 = fmaxf(bx1, x1), yy1 = fmaxf(by1, y1);
                    float xx2 = fminf(bx2, x2), yy2 = fminf(by2, y2);
                    float inter = fmaxf(xx2 - xx1 + 1.f, 0.f) * fmaxf(yy2 - yy1 + 1.f, 0.f);
                    float area = (x2 - x1 + 1.f) * (y2 - y1 + 1.f);
                    float iou = inter / (barea + area - inter);
                    if (iou > 0.7f) nv &= ~(1u << b);
                }
                valid[t] = nv;
            }
        }
        __syncthreads();
    }
}

__global__ void k_make_rois(const float* __restrict__ props, const int* __restrict__ keep,
                            float* __restrict__ rois) {
    int r = blockIdx.x * 256 + threadIdx.x;
    if (r >= POST_N) return;
    int k = keep[r];
    rois[r * 5 + 0] = 0.f;
    rois[r * 5 + 1] = props[k * 4 + 0];
    rois[r * 5 + 2] = props[k * 4 + 1];
    rois[r * 5 + 3] = props[k * 4 + 2];
    rois[r * 5 + 4] = props[k * 4 + 3];
}

// CHW -> HWC so ROI-crop channel gathers are coalesced.
__global__ void k_hwc(const float* __restrict__ f, float* __restrict__ fhwc) {
    int i = blockIdx.x * 256 + threadIdx.x;   // CIN*NPIX
    int c = i / NPIX, p = i - c * NPIX;
    fhwc[p * CIN + c] = f[i];
}

__global__ __launch_bounds__(256) void k_roi_crop(const float* __restrict__ fhwc,
                                                  const float* __restrict__ rois,
                                                  float* __restrict__ out) {
    int r = blockIdx.x;
    __shared__ int   sidx[49][4];
    __shared__ float swgt[49][4];
    int t = threadIdx.x;
    if (t < 49) {
        int py = t / 7, px = t - py * 7;
        float x1 = rois[r * 5 + 1] / 16.f, y1 = rois[r * 5 + 2] / 16.f;
        float x2 = rois[r * 5 + 3] / 16.f, y2 = rois[r * 5 + 4] / 16.f;
        float t00 = (x2 - x1) / 95.f, t02 = (x1 + x2 - 95.f) / 95.f;
        float t11 = (y2 - y1) / 95.f, t12 = (y1 + y2 - 95.f) / 95.f;
        float lx = (px - 3) / 3.f, ly = (py - 3) / 3.f;
        float gx = t00 * lx + t02, gy = t11 * ly + t12;
        float ix = (gx + 1.f) * 0.5f * 95.f, iy = (gy + 1.f) * 0.5f * 95.f;
        float ix0 = floorf(ix), iy0 = floorf(iy);
        float wx = ix - ix0, wy = iy - iy0;
        float ys[2]  = {iy0, iy0 + 1.f};
        float xs[2]  = {ix0, ix0 + 1.f};
        float wys[2] = {1.f - wy, wy};
        float wxs[2] = {1.f - wx, wx};
        for (int a = 0; a < 2; ++a)
            for (int b = 0; b < 2; ++b) {
                float yi = ys[a], xi = xs[b];
                bool valid = (yi >= 0.f) && (yi < 96.f) && (xi >= 0.f) && (xi < 96.f);
                int yc = (int)fminf(fmaxf(yi, 0.f), 95.f);
                int xc = (int)fminf(fmaxf(xi, 0.f), 95.f);
                sidx[t][a * 2 + b] = yc * FW + xc;
                swgt[t][a * 2 + b] = wys[a] * wxs[b] * (valid ? 1.f : 0.f);
            }
    }
    __syncthreads();
    for (int c = t; c < CIN; c += 256) {
        for (int k = 0; k < 49; ++k) {
            float v = swgt[k][0] * fhwc[sidx[k][0] * CIN + c]
                    + swgt[k][1] * fhwc[sidx[k][1] * CIN + c]
                    + swgt[k][2] * fhwc[sidx[k][2] * CIN + c]
                    + swgt[k][3] * fhwc[sidx[k][3] * CIN + c];
            out[((size_t)r * CIN + c) * 49 + k] = v;
        }
    }
}

// ---------------------------------------------------------------------------
// Anchor targets.
// ---------------------------------------------------------------------------
__device__ __forceinline__ float iou_fn(float ax1, float ay1, float ax2, float ay2,
                                        float bx1, float by1, float bx2, float by2) {
    float aa = (ax2 - ax1 + 1.f) * (ay2 - ay1 + 1.f);
    float ba = (bx2 - bx1 + 1.f) * (by2 - by1 + 1.f);
    float x1 = fmaxf(ax1, bx1), y1 = fmaxf(ay1, by1);
    float x2 = fminf(ax2, bx2), y2 = fminf(ay2, by2);
    float inter = fmaxf(x2 - x1 + 1.f, 0.f) * fmaxf(y2 - y1 + 1.f, 0.f);
    return inter / (aa + ba - inter);
}

__global__ __launch_bounds__(256) void k_gtmax(const float* __restrict__ anch,
                                               const float* __restrict__ gt,
                                               const float* __restrict__ iminfo,
                                               float* __restrict__ gtmax) {
    int g = blockIdx.x;
    float gx1 = gt[g * 5 + 0], gy1 = gt[g * 5 + 1];
    float gx2 = gt[g * 5 + 2], gy2 = gt[g * 5 + 3];
    float imh = iminfo[0], imw = iminfo[1];
    float mx = -1e30f;
    for (int i = threadIdx.x; i < NANCH; i += 256) {
        float ax1 = anch[i * 4], ay1 = anch[i * 4 + 1];
        float ax2 = anch[i * 4 + 2], ay2 = anch[i * 4 + 3];
        bool inside = (ax1 >= 0.f) && (ay1 >= 0.f) && (ax2 < imw) && (ay2 < imh);
        float ov = inside ? iou_fn(ax1, ay1, ax2, ay2, gx1, gy1, gx2, gy2) : -1.f;
        mx = fmaxf(mx, ov);
    }
    __shared__ float sm[256];
    sm[threadIdx.x] = mx; __syncthreads();
    for (int s = 128; s > 0; s >>= 1) {
        if (threadIdx.x < s) sm[threadIdx.x] = fmaxf(sm[threadIdx.x], sm[threadIdx.x + s]);
        __syncthreads();
    }
    if (threadIdx.x == 0) gtmax[g] = sm[0];
}

__global__ void k_zero(int* cnt) { if (threadIdx.x == 0 && blockIdx.x == 0) cnt[0] = 0; }

__global__ void k_targets(const float* __restrict__ anch, const float* __restrict__ gt,
                          const float* __restrict__ iminfo, const float* __restrict__ gtmax,
                          float* __restrict__ labels, float* __restrict__ tgt,
                          int* __restrict__ cnt, int G) {
    int i = blockIdx.x * 256 + threadIdx.x;
    float ax1 = anch[i * 4], ay1 = anch[i * 4 + 1];
    float ax2 = anch[i * 4 + 2], ay2 = anch[i * 4 + 3];
    float imh = iminfo[0], imw = iminfo[1];
    bool inside = (ax1 >= 0.f) && (ay1 >= 0.f) && (ax2 < imw) && (ay2 < imh);
    float maxov = -1e30f; int am = 0; bool best = false;
    for (int g = 0; g < G; ++g) {
        float ov = inside ? iou_fn(ax1, ay1, ax2, ay2,
                                   gt[g * 5], gt[g * 5 + 1], gt[g * 5 + 2], gt[g * 5 + 3])
                          : -1.f;
        if (ov > maxov) { maxov = ov; am = g; }
        if (ov == gtmax[g]) best = true;
    }
    float lab = -1.f;
    if (inside && maxov < 0.3f) lab = 0.f;
    if (inside && best) lab = 1.f;
    if (inside && maxov >= 0.7f) lab = 1.f;
    labels[i] = lab;
    float gx1 = gt[am * 5], gy1 = gt[am * 5 + 1], gx2 = gt[am * 5 + 2], gy2 = gt[am * 5 + 3];
    float ew = ax2 - ax1 + 1.f, eh = ay2 - ay1 + 1.f;
    float ecx = ax1 + 0.5f * ew, ecy = ay1 + 0.5f * eh;
    float gw = gx2 - gx1 + 1.f, gh = gy2 - gy1 + 1.f;
    float gcx = gx1 + 0.5f * gw, gcy = gy1 + 0.5f * gh;
    tgt[i * 4 + 0] = (gcx - ecx) / ew;
    tgt[i * 4 + 1] = (gcy - ecy) / eh;
    tgt[i * 4 + 2] = logf(gw / ew);
    tgt[i * 4 + 3] = logf(gh / eh);
    if (lab >= 0.f) atomicAdd(cnt, 1);   // integer atomic: deterministic
}

// ---------------------------------------------------------------------------
// Loss partials (deterministic block-tree reductions, no float atomics).
// ---------------------------------------------------------------------------
__global__ __launch_bounds__(256) void k_cls_partial(const float* __restrict__ cls,
                                                     const float* __restrict__ labels,
                                                     float* __restrict__ pce,
                                                     float* __restrict__ pmask) {
    int i = blockIdx.x * 256 + threadIdx.x;
    int p = i / NA, a = i - p * NA;
    float lab = labels[i];
    float mask = (lab >= 0.f) ? 1.f : 0.f;
    float l0 = cls[a * NPIX + p], l1 = cls[(NA + a) * NPIX + p];
    float mx = fmaxf(l0, l1);
    float lse = mx + logf(expf(l0 - mx) + expf(l1 - mx));
    float chosen = (lab >= 1.f) ? l1 : l0;
    float ce = -(chosen - lse);
    __shared__ float s1[256], s2[256];
    int t = threadIdx.x;
    s1[t] = ce * mask; s2[t] = mask; __syncthreads();
    for (int s = 128; s > 0; s >>= 1) {
        if (t < s) { s1[t] += s1[t + s]; s2[t] += s2[t + s]; }
        __syncthreads();
    }
    if (t == 0) { pce[blockIdx.x] = s1[0]; pmask[blockIdx.x] = s2[0]; }
}

__global__ __launch_bounds__(256) void k_box_partial(const float* __restrict__ bbox,
                                                     const float* __restrict__ tgt,
                                                     const float* __restrict__ labels,
                                                     const int* __restrict__ cnt,
                                                     float* __restrict__ pbox) {
    int i = blockIdx.x * 256 + threadIdx.x;
    int p = i / NA, a = i - p * NA;
    float lab = labels[i];
    float iw = (lab == 1.f) ? 1.f : 0.f;
    float ex = (lab >= 0.f) ? 1.f : 0.f;
    float nex = fmaxf((float)cnt[0], 1.f);
    float s = 0.f;
    #pragma unroll
    for (int j = 0; j < 4; ++j) {
        float d  = bbox[(a * 4 + j) * NPIX + p];
        float df = iw * (d - tgt[i * 4 + j]);
        float ad = fabsf(df);
        float sl1 = (ad < 1.f / 9.f) ? (0.5f * 9.f * df * df) : (ad - 0.5f / 9.f);
        s += sl1;
    }
    float v = (ex / nex) * s;
    __shared__ float sm[256];
    int t = threadIdx.x;
    sm[t] = v; __syncthreads();
    for (int st = 128; st > 0; st >>= 1) {
        if (t < st) sm[t] += sm[t + st];
        __syncthreads();
    }
    if (t == 0) pbox[blockIdx.x] = sm[0];
}

__global__ void k_finalize(const float* __restrict__ pce, const float* __restrict__ pmask,
                           const float* __restrict__ pbox, float* __restrict__ losses) {
    if (threadIdx.x == 0 && blockIdx.x == 0) {
        float sce = 0.f, sm = 0.f, sb = 0.f;
        for (int i = 0; i < NANCH / 256; ++i) { sce += pce[i]; sm += pmask[i]; sb += pbox[i]; }
        losses[0] = sce / fmaxf(sm, 1.f);
        losses[1] = sb;
    }
}

// ---------------------------------------------------------------------------
extern "C" void kernel_launch(void* const* d_in, const int* in_sizes, int n_in,
                              void* d_out, int out_size, void* d_ws, size_t ws_size,
                              hipStream_t stream) {
    const float* feat   = (const float*)d_in[0];
    const float* gt     = (const float*)d_in[1];
    const float* iminfo = (const float*)d_in[2];
    const float* rpn_w  = (const float*)d_in[3];
    const float* rpn_b  = (const float*)d_in[4];
    const float* cls_w  = (const float*)d_in[5];
    const float* cls_b  = (const float*)d_in[6];
    const float* bbox_w = (const float*)d_in[7];
    const float* bbox_b = (const float*)d_in[8];
    const int G = in_sizes[1] / 5;

    char* ws = (char*)d_ws;
    size_t off = 0;
    auto carve = [&](size_t bytes) -> void* {
        void* p = ws + off;
        off += (bytes + 255) & ~(size_t)255;
        return p;
    };
    _Float16* wpk   = (_Float16*)carve((size_t)9 * CMID * CIN * 2);
    _Float16* rpnh  = (_Float16*)carve((size_t)CMID * NPIX * 2);
    float* cls_out  = (float*)carve((size_t)18 * NPIX * 4);
    float* bbox_out = (float*)carve((size_t)36 * NPIX * 4);
    float* scores   = (float*)carve((size_t)NANCH * 4);
    float* anch     = (float*)carve((size_t)NANCH * 4 * 4);
    float* prop     = (float*)carve((size_t)NANCH * 4 * 4);
    unsigned long long* keys = (unsigned long long*)carve((size_t)SORT_N * 8);
    float* props    = (float*)carve((size_t)PRE_N * 4 * 4);
    int*   keep     = (int*)carve((size_t)POST_N * 4);
    float* rois     = (float*)carve((size_t)POST_N * 5 * 4);
    float* fhwc     = (float*)carve((size_t)NPIX * CIN * 4);
    float* labels   = (float*)carve((size_t)NANCH * 4);
    float* tgt      = (float*)carve((size_t)NANCH * 4 * 4);
    float* gtmax    = (float*)carve(64);
    int*   cnt      = (int*)carve(64);
    float* pce      = (float*)carve((NANCH / 256) * 4);
    float* pmask    = (float*)carve((NANCH / 256) * 4);
    float* pbox     = (float*)carve((NANCH / 256) * 4);

    float* pooled = (float*)d_out;
    float* losses = pooled + (size_t)POST_N * CIN * POOL * POOL;

    k_pack_w<<<(9 * CMID * CIN + 255) / 256, 256, 0, stream>>>(rpn_w, wpk);
    k_conv3x3<<<dim3(FH, FW / 32, CMID / 256), 256, 0, stream>>>(feat, wpk, rpn_b, rpnh);
    k_conv1x1<<<dim3(NPIX / 256, 18), 256, 0, stream>>>(rpnh, cls_w, cls_b, cls_out);
    k_conv1x1<<<dim3(NPIX / 256, 36), 256, 0, stream>>>(rpnh, bbox_w, bbox_b, bbox_out);
    k_scores<<<NANCH / 256, 256, 0, stream>>>(cls_out, scores);
    k_anch_decode<<<NANCH / 256, 256, 0, stream>>>(bbox_out, iminfo, anch, prop);
    k_init_keys<<<SORT_N / 256, 256, 0, stream>>>(scores, keys);
    for (int k = 2; k <= SORT_N; k <<= 1)
        for (int j = k >> 1; j > 0; j >>= 1)
            k_bitonic<<<SORT_N / 256, 256, 0, stream>>>(keys, k, j);
    k_gather_top<<<(PRE_N + 255) / 256, 256, 0, stream>>>(keys, prop, props);
    k_nms<<<1, 512, 0, stream>>>(props, keep);
    k_make_rois<<<(POST_N + 255) / 256, 256, 0, stream>>>(props, keep, rois);
    k_hwc<<<(CIN * NPIX) / 256, 256, 0, stream>>>(feat, fhwc);
    k_roi_crop<<<POST_N, 256, 0, stream>>>(fhwc, rois, pooled);
    k_zero<<<1, 1, 0, stream>>>(cnt);
    k_gtmax<<<G, 256, 0, stream>>>(anch, gt, iminfo, gtmax);
    k_targets<<<NANCH / 256, 256, 0, stream>>>(anch, gt, iminfo, gtmax, labels, tgt, cnt, G);
    k_cls_partial<<<NANCH / 256, 256, 0, stream>>>(cls_out, labels, pce, pmask);
    k_box_partial<<<NANCH / 256, 256, 0, stream>>>(bbox_out, tgt, labels, cnt, pbox);
    k_finalize<<<1, 1, 0, stream>>>(pce, pmask, pbox, losses);
}
